// MemoryEmbedding_39221641347613
// MI455X (gfx1250) — matrix operand order repaired and checked by reading, not compiled, not run on hardware
//
#include <hip/hip_runtime.h>
#include <hip/hip_bf16.h>

// Problem constants (match reference)
#define OBS_DIM 64
#define EMB     128
#define HID     256
#define G4      1024   // 4*HID
#define BATCH   32
#define TLEN    2048

typedef __attribute__((ext_vector_type(16))) __bf16 v16bf;
typedef __attribute__((ext_vector_type(8)))  float  v8f;
// Global-address-space fragment pointer: forces global_load_b128 (LOADcnt only),
// not flat_load_b128 (which also ties up DScnt / the LDS path).
typedef __attribute__((address_space(1))) const v16bf* gfrag_t;

// ---- WMMA fragment layout helpers (halves within a 512-half, 32B/lane tile) ----
// A-matrix 16x32 bf16 (ISA 7.12.2): lane<16 holds row=lane, K in {0..7,16..23};
// lane>=16 holds row=lane-16, K in {8..15,24..31}; packed pairs per VGPR.
__device__ __forceinline__ int pack_offA(int kk, int row) {
    int hi     = ((kk & 15) >= 8) ? 1 : 0;
    int e      = ((kk >> 4) << 3) | (kk & 7);
    int lane   = row + (hi << 4);
    return lane * 16 + e;
}
// B-matrix 32x16 bf16: lanes 0-15 = cols with K in {0..7,16..23}, lanes 16-31 = K in {8..15,24..31}.
__device__ __forceinline__ int pack_offB(int kk, int col) {
    int hi   = ((kk & 15) >= 8) ? 1 : 0;
    int e    = ((kk >> 4) << 3) | (kk & 7);
    int lane = col + (hi << 4);
    return lane * 16 + e;
}

__device__ __forceinline__ unsigned short f2bf(float f) {
    __bf16 h = (__bf16)f;
    return __builtin_bit_cast(unsigned short, h);
}

// ---- Prep kernel: rec_kernel [256,1024] f32 -> bf16 B-fragment tiles [kt8][nt64][lane32][16h]
__global__ void pack_rec_kernel(const float* __restrict__ rec, unsigned short* __restrict__ outp) {
    int gid = blockIdx.x * blockDim.x + threadIdx.x;      // 262144
    int k = gid >> 10, j = gid & 1023;
    int kt = k >> 5, kk = k & 31, nt = j >> 4, col = j & 15;
    outp[(kt * 64 + nt) * 512 + pack_offB(kk, col)] = f2bf(rec[gid]);
}

// ---- Prep kernel: Wc = W_embed[64,128] @ kernel[128,1024] -> bf16 B tiles [kt2][nt64][lane32][16h]
__global__ void build_wc_kernel(const float* __restrict__ W_embed, const float* __restrict__ kern,
                                unsigned short* __restrict__ outp) {
    int gid = blockIdx.x * blockDim.x + threadIdx.x;      // 65536
    int i = gid >> 10, j = gid & 1023;
    float s = 0.f;
    #pragma unroll 8
    for (int k = 0; k < EMB; ++k) s += W_embed[i * EMB + k] * kern[k * G4 + j];
    int kt = i >> 5, kk = i & 31, nt = j >> 4, col = j & 15;
    outp[(kt * 64 + nt) * 512 + pack_offB(kk, col)] = f2bf(s);
}

// ---- Prep kernel: bc = b_embed @ kernel + bias  (f32, 1024)
__global__ void build_bc_kernel(const float* __restrict__ b_embed, const float* __restrict__ kern,
                                const float* __restrict__ bias, float* __restrict__ bc) {
    int j = threadIdx.x;
    float s = bias[j];
    #pragma unroll 8
    for (int k = 0; k < EMB; ++k) s += b_embed[k] * kern[k * G4 + j];
    bc[j] = s;
}

// ---- Main recurrent kernel: 2 blocks x 1024 threads (32 waves), 16 batch rows per block.
__global__ __launch_bounds__(1024, 1)
void lstm_scan_kernel(const float* __restrict__ obs, const float* __restrict__ S,
                      const float* __restrict__ M,
                      const unsigned short* __restrict__ rec_p,
                      const unsigned short* __restrict__ wc_p,
                      const float* __restrict__ bc_g, float* __restrict__ out) {
    __shared__ __align__(32) unsigned short hA[8 * 512];  // h bf16, A-fragment layout (8 k-tiles)
    __shared__ __align__(32) unsigned short xA[2 * 512];  // obs_t bf16, A-fragment layout
    __shared__ float zS[16 * G4];                         // z = x@Wc + h@rec  (64 KB)
    __shared__ float cS[16 * HID];                        // cell state f32
    __shared__ float bcS[G4];

    const int tid  = threadIdx.x;
    const int blk  = blockIdx.x;       // 0..1
    const int lane = tid & 31;
    const int w    = tid >> 5;         // wave 0..31
    const int b0   = blk * 16;

    bcS[tid] = bc_g[tid];
    // initial [h,c] from S
    #pragma unroll
    for (int i = 0; i < 4; ++i) {
        int lin = tid * 4 + i;
        int bl = lin >> 8, j = lin & 255;
        float h0 = S[(b0 + bl) * (2 * HID) + j];
        float c0 = S[(b0 + bl) * (2 * HID) + HID + j];
        cS[bl * HID + j] = c0;
        hA[(j >> 5) * 512 + pack_offA(j & 31, bl)] = f2bf(h0);
    }
    __syncthreads();

    const int nt0 = w * 2, nt1 = w * 2 + 1;

    #pragma clang loop unroll(disable)
    for (int t = 0; t < TLEN; ++t) {
        // Launder weight base pointers each iteration: forces the B-fragment
        // loads to stay inside the loop (L2-resident re-fetch) instead of being
        // hoisted into VGPRs and spilled to scratch. Then cast back into the
        // global address space so they lower as global_load_b128.
        const unsigned short* rec_it = rec_p;
        const unsigned short* wc_it  = wc_p;
        asm volatile("" : "+s"(rec_it), "+s"(wc_it));
        gfrag_t recF = (gfrag_t)(const v16bf*)rec_it;
        gfrag_t wcF  = (gfrag_t)(const v16bf*)wc_it;

        // ---- phase 0: episode-reset mask + stage obs_t into A-fragments ----
        {
            int lin = tid * 4;
            int bl = lin >> 8, j = lin & 255;
            float m = M[(b0 + bl) * TLEN + t];
            if (m != 0.0f) {
                #pragma unroll
                for (int i = 0; i < 4; ++i) {
                    int jj = j + i;
                    cS[bl * HID + jj] = 0.0f;
                    hA[(jj >> 5) * 512 + pack_offA(jj & 31, bl)] = 0;
                }
            }
            int bo = tid >> 6, k = tid & 63;   // one obs element per thread
            float ov = obs[((size_t)(b0 + bo) * TLEN + t) * OBS_DIM + k];
            xA[(k >> 5) * 512 + pack_offA(k & 31, bo)] = f2bf(ov);
        }
        __syncthreads();

        // ---- phase 1: z = obs_t@Wc + h@rec via bf16 WMMA, f32 accum ----
        v8f acc0 = {}; v8f acc1 = {};
        #pragma unroll
        for (int kt = 0; kt < 2; ++kt) {
            v16bf a  = *reinterpret_cast<const v16bf*>(&xA[kt * 512 + lane * 16]);
            v16bf bA = wcF[(kt * 64 + nt0) * 32 + lane];
            v16bf bB = wcF[(kt * 64 + nt1) * 32 + lane];
            acc0 = __builtin_amdgcn_wmma_f32_16x16x32_bf16(false, a, false, bA, (short)0, acc0, false, false);
            acc1 = __builtin_amdgcn_wmma_f32_16x16x32_bf16(false, a, false, bB, (short)0, acc1, false, false);
        }
        #pragma unroll
        for (int kt = 0; kt < 8; ++kt) {
            v16bf a  = *reinterpret_cast<const v16bf*>(&hA[kt * 512 + lane * 16]);
            v16bf bA = recF[(kt * 64 + nt0) * 32 + lane];
            v16bf bB = recF[(kt * 64 + nt1) * 32 + lane];
            acc0 = __builtin_amdgcn_wmma_f32_16x16x32_bf16(false, a, false, bA, (short)0, acc0, false, false);
            acc1 = __builtin_amdgcn_wmma_f32_16x16x32_bf16(false, a, false, bB, (short)0, acc1, false, false);
        }
        {   // spill D tiles to LDS (C/D layout: vgpr e -> row e (+8 for upper lanes), col = lane&15)
            int rhi = (lane >= 16) ? 8 : 0;
            int col = lane & 15;
            #pragma unroll
            for (int e = 0; e < 8; ++e) {
                zS[(e + rhi) * G4 + nt0 * 16 + col] = acc0[e];
                zS[(e + rhi) * G4 + nt1 * 16 + col] = acc1[e];
            }
        }
        __syncthreads();

        // ---- phase 2: gates (keras order i,f,g,o), state update, output ----
        #pragma unroll
        for (int i = 0; i < 4; ++i) {
            int lin = tid * 4 + i;
            int bl = lin >> 8, j = lin & 255;
            float zi = zS[bl * G4 + j]       + bcS[j];
            float zf = zS[bl * G4 + 256 + j] + bcS[256 + j];
            float zg = zS[bl * G4 + 512 + j] + bcS[512 + j];
            float zo = zS[bl * G4 + 768 + j] + bcS[768 + j];
            float ig = 1.0f / (1.0f + __expf(-zi));
            float fg = 1.0f / (1.0f + __expf(-zf));
            float gg = tanhf(zg);
            float og = 1.0f / (1.0f + __expf(-zo));
            float c  = fg * cS[bl * HID + j] + ig * gg;
            float h  = og * tanhf(c);
            cS[bl * HID + j] = c;
            __builtin_nontemporal_store(h, &out[((size_t)(b0 + bl) * TLEN + t) * HID + j]);
            hA[(j >> 5) * 512 + pack_offA(j & 31, bl)] = f2bf(h);
        }
        __syncthreads();
    }
}

extern "C" void kernel_launch(void* const* d_in, const int* in_sizes, int n_in,
                              void* d_out, int out_size, void* d_ws, size_t ws_size,
                              hipStream_t stream) {
    const float* obs        = (const float*)d_in[0];
    const float* S          = (const float*)d_in[1];
    const float* M          = (const float*)d_in[2];
    const float* W_embed    = (const float*)d_in[3];
    const float* b_embed    = (const float*)d_in[4];
    const float* kern       = (const float*)d_in[5];
    const float* rec_kernel = (const float*)d_in[6];
    const float* bias       = (const float*)d_in[7];
    float* out = (float*)d_out;

    // workspace layout: rec bf16 (512KB) | Wc bf16 (128KB) | bc f32 (4KB)
    unsigned short* rec_p = (unsigned short*)d_ws;
    unsigned short* wc_p  = rec_p + (HID * G4);              // 262144 halves
    float*          bc    = (float*)(wc_p + (OBS_DIM * G4)); // 65536 halves

    pack_rec_kernel<<<(HID * G4) / 1024, 1024, 0, stream>>>(rec_kernel, rec_p);
    build_wc_kernel<<<(OBS_DIM * G4) / 1024, 1024, 0, stream>>>(W_embed, kern, wc_p);
    build_bc_kernel<<<1, 1024, 0, stream>>>(b_embed, kern, bias, bc);
    lstm_scan_kernel<<<2, 1024, 0, stream>>>(obs, S, M, rec_p, wc_p, bc, out);
}
